// TargetUnitsHead_36713380446757
// MI455X (gfx1250) — compile-verified, matrix operand order checked by reading, not run on hardware
//
#include <hip/hip_runtime.h>
#include <hip/hip_bf16.h>

// Problem dims (match reference)
#define B_    256
#define N_    512
#define ED_   256
#define ID_   1024
#define UTD_  256
#define FD_   256
#define K_    64
#define H_    64
#define T_    64
#define NP1   513   // N+1 (end slot appended)

typedef __attribute__((ext_vector_type(16))) __bf16 v16bf;
typedef __attribute__((ext_vector_type(8)))  float  v8f;

// 16-bit A/B operand K-index pattern for v_wmma_*_16x16x32_* (CDNA5 ISA §7.12.2):
// lanes 0-15 hold K=0..7 in vgprs0-3 (pairs) and K=16..23 in vgprs4-7;
// lanes 16-31 hold K=8..15 and K=24..31.
__device__ __forceinline__ int kb_pattern(int v, int half) {
  return (v < 4) ? (2 * v + 8 * half) : (16 + 2 * (v - 4) + 8 * half);
}

// ---------------------------------------------------------------------------
// Kernel 0: swizzle Wk (f32 [ED,K]) into per-lane-contiguous bf16 B operands,
// and zero the appended end-slot row of key. 64 blocks x 256 threads.
// packedB index: ((cb*8 + ch)*32 + lane)*16 + e  (32 bytes per lane per WMMA)
// ---------------------------------------------------------------------------
__global__ __launch_bounds__(256) void pack_b_kernel(
    const float* __restrict__ Wk, __bf16* __restrict__ packedB,
    float* __restrict__ keyout) {
  int gid = blockIdx.x * blockDim.x + threadIdx.x;   // 16384 threads
  // zero end-slot row key[b, N_, :]
  int b = gid / K_, k = gid % K_;
  keyout[((size_t)b * NP1 + N_) * K_ + k] = 0.0f;

  if (gid < 1024) {                                  // 4 cb * 8 ch * 32 lanes
    int lane = gid & 31;
    int ch   = (gid >> 5) & 7;
    int cb   = gid >> 8;
    int half = lane >> 4, n = lane & 15;
    __bf16* dst = packedB + (size_t)gid * 16;
#pragma unroll
    for (int v = 0; v < 8; ++v) {
      int kk = kb_pattern(v, half);
      dst[2 * v]     = (__bf16)Wk[(size_t)(ch * 32 + kk)     * K_ + cb * 16 + n];
      dst[2 * v + 1] = (__bf16)Wk[(size_t)(ch * 32 + kk + 1) * K_ + cb * 16 + n];
    }
  }
}

// ---------------------------------------------------------------------------
// Kernel 1: key = ee @ Wk + bk  via v_wmma_f32_16x16x32_bf16.
// M = B*N = 131072 rows, Kdim = 256, Ndim = 64.
// 1 wave -> one 16-row tile, all 4 col-blocks; 32 WMMAs per wave.
// 1024 blocks x 256 threads (8 waves).
// ---------------------------------------------------------------------------
__global__ __launch_bounds__(256) void key_gemm_kernel(
    const float* __restrict__ ee, const float* __restrict__ bk,
    const __bf16* __restrict__ packedB, float* __restrict__ keyout) {
  const int lane = threadIdx.x & 31;
  const int wave = threadIdx.x >> 5;
  const int half = lane >> 4, m = lane & 15;
  const size_t row0 = ((size_t)blockIdx.x * 8 + wave) * 16;
  const float* arow = ee + (row0 + m) * ED_;

  // Preload the full 16x256 A tile as bf16 (8 x v16bf = 64 VGPRs)
  v16bf a[8];
#pragma unroll
  for (int ch = 0; ch < 8; ++ch) {
#pragma unroll
    for (int v = 0; v < 8; ++v) {
      int kk = ch * 32 + kb_pattern(v, half);
      a[ch][2 * v]     = (__bf16)arow[kk];
      a[ch][2 * v + 1] = (__bf16)arow[kk + 1];
    }
  }

#pragma unroll
  for (int cb = 0; cb < 4; ++cb) {
    v8f acc = {};
#pragma unroll
    for (int ch = 0; ch < 8; ++ch) {
      v16bf bm = *(const v16bf*)(packedB + ((size_t)(cb * 8 + ch) * 32 + lane) * 16);
      acc = __builtin_amdgcn_wmma_f32_16x16x32_bf16(
          /*neg_a=*/false, a[ch], /*neg_b=*/false, bm,
          /*c_mod=*/(short)0, acc, /*reuse_a=*/false, /*reuse_b=*/false);
    }
    const int col = cb * 16 + m;
    const float bias = bk[col];
#pragma unroll
    for (int r = 0; r < 8; ++r) {
      size_t grow = row0 + (size_t)(r + 8 * half);   // C/D layout: vgpr r -> M=r(+8)
      size_t bb = grow >> 9;                          // /512
      size_t nn = grow & 511;
      keyout[(bb * NP1 + nn) * K_ + col] = acc[r] + bias;
    }
  }
}

// ---------------------------------------------------------------------------
// Kernel 2: fused init-query MLP.
// func = relu(autm@Wf+bf); x1 = relu(emb@W1+b1); x0 = relu((x1+func)@W2+b2)
// 256 blocks (one per batch) x 256 threads.
// ---------------------------------------------------------------------------
__global__ __launch_bounds__(256) void init_query_kernel(
    const float* __restrict__ emb,  const float* __restrict__ autm,
    const float* __restrict__ Wf,   const float* __restrict__ bfv,
    const float* __restrict__ W1,   const float* __restrict__ b1,
    const float* __restrict__ W2,   const float* __restrict__ b2,
    float* __restrict__ x0out) {
  __shared__ float s[FD_];
  const int b = blockIdx.x, j = threadIdx.x;
  float acc1 = bfv[j];
  for (int u = 0; u < UTD_; ++u)
    acc1 = fmaf(autm[b * UTD_ + u], Wf[u * FD_ + j], acc1);
  float func = fmaxf(acc1, 0.f);
  float acc2 = b1[j];
  for (int i = 0; i < ID_; ++i)
    acc2 = fmaf(emb[b * ID_ + i], W1[i * FD_ + j], acc2);
  float x1 = fmaxf(acc2, 0.f);
  s[j] = x1 + func;
  __syncthreads();
  if (j < K_) {
    float acc = b2[j];
    for (int q = 0; q < FD_; ++q) acc = fmaf(s[q], W2[q * K_ + j], acc);
    x0out[b * K_ + j] = fmaxf(acc, 0.f);
  }
}

// ---------------------------------------------------------------------------
// Kernel 3: persistent per-batch LN-LSTM + masked-argmax selection scan.
// One block per batch element; key[b] (513x64 f32 = 131KB) lives in LDS
// (CDNA5: 320KB/WGP), avoiding 2.1GB of HBM re-reads across the 64 steps.
// ---------------------------------------------------------------------------
__device__ __forceinline__ float sigf(float x) { return 1.0f / (1.0f + expf(-x)); }

__device__ __forceinline__ float block_sum(float* red, float val, int t) {
  red[t] = val; __syncthreads();
#pragma unroll
  for (int s2 = 128; s2 > 0; s2 >>= 1) {
    if (t < s2) red[t] += red[t + s2];
    __syncthreads();
  }
  float r = red[0];
  __syncthreads();
  return r;
}

__global__ __launch_bounds__(256) void lstm_select_kernel(
    const float* __restrict__ keybuf, const float* __restrict__ x0buf,
    const float* __restrict__ aum,    const int* __restrict__ oen,
    const float* __restrict__ wx,     const float* __restrict__ wh,
    const float* __restrict__ lstm_b,
    const float* __restrict__ gxg, const float* __restrict__ gxb,
    const float* __restrict__ ghg, const float* __restrict__ ghb,
    float* __restrict__ logits, int* __restrict__ units) {
  extern __shared__ float sm[];
  float* keys = sm;                       // NP1*K_
  float* hv   = keys + NP1 * K_;          // 64
  float* cv   = hv + H_;                  // 64
  float* xv   = cv + H_;                  // 64
  float* gu   = xv + H_;                  // 256 (x@wx pre-act, then final gates)
  float* gvv  = gu + 4 * H_;              // 256 (h@wh pre-act)
  float* red  = gvv + 4 * H_;             // 256 reduction scratch
  float* mask = red + 256;                // 513
  int*   redi = (int*)(mask + NP1);       // 256

  const int t = threadIdx.x;
  const int b = blockIdx.x;

  for (int i = t; i < NP1 * K_; i += 256) keys[i] = keybuf[(size_t)b * NP1 * K_ + i];
  for (int n = t; n < NP1; n += 256) {
    mask[n] = (n < N_) ? aum[b * N_ + n] : 1.0f;
    units[b * NP1 + n] = 0;
  }
  if (t < H_) { xv[t] = x0buf[b * H_ + t]; hv[t] = 0.f; cv[t] = 0.f; }
  __syncthreads();

  const int myoen = oen[b];

  for (int step = 0; step < T_; ++step) {
    // gate pre-activations: u = (x @ wx)[t], v = (h @ wh)[t]
    float u = 0.f, v = 0.f;
    for (int k = 0; k < H_; ++k) {
      u = fmaf(xv[k], wx[k * (4 * H_) + t], u);
      v = fmaf(hv[k], wh[k * (4 * H_) + t], v);
    }
    gu[t] = u; gvv[t] = v;
    __syncthreads();

    // LayerNorm over the 4H=256 gate dimension for each stream
    float mu_u = block_sum(red, u, t) * (1.0f / 256.0f);
    float var_u = block_sum(red, (u - mu_u) * (u - mu_u), t) * (1.0f / 256.0f);
    float mu_v = block_sum(red, v, t) * (1.0f / 256.0f);
    float var_v = block_sum(red, (v - mu_v) * (v - mu_v), t) * (1.0f / 256.0f);

    float g = (u - mu_u) * rsqrtf(var_u + 1e-5f) * gxg[t] + gxb[t]
            + (v - mu_v) * rsqrtf(var_v + 1e-5f) * ghg[t] + ghb[t]
            + lstm_b[t];
    gu[t] = g;
    __syncthreads();

    // LSTM cell update (gate order i,f,o,u)
    if (t < H_) {
      float gi = gu[t], gf = gu[H_ + t], go = gu[2 * H_ + t], gc = gu[3 * H_ + t];
      float c = sigf(gf) * cv[t] + sigf(gi) * tanhf(gc);
      float h = sigf(go) * tanhf(c);
      cv[t] = c; hv[t] = h; xv[t] = h;   // LSTM output fed back as next input
    }
    __syncthreads();

    // qr[n] = mean_k(h[k]*key[n,k]) - (1-mask[n])*1e9 ; local first-index argmax
    float bv = -3.4e38f; int bi = 0;
    for (int n = t; n < NP1; n += 256) {
      const float* kr = keys + n * K_;
      float acc = 0.f;
#pragma unroll 8
      for (int k = 0; k < K_; ++k) acc = fmaf(hv[k], kr[k], acc);
      float qr = acc * (1.0f / 64.0f) - (1.0f - mask[n]) * 1e9f;
      logits[((size_t)step * B_ + b) * NP1 + n] = qr;
      if (qr > bv) { bv = qr; bi = n; }     // strict > keeps lowest index
    }
    red[t] = bv; redi[t] = bi;
    __syncthreads();
#pragma unroll
    for (int s2 = 128; s2 > 0; s2 >>= 1) {
      if (t < s2) {
        float ov = red[t + s2]; int oi = redi[t + s2];
        if (ov > red[t] || (ov == red[t] && oi < redi[t])) { red[t] = ov; redi[t] = oi; }
      }
      __syncthreads();
    }

    // selection / mask update (argmax(softmax(qr/temp)) == argmax(qr))
    if (t == 0 && step < myoen) {
      int sel = redi[0];
      units[b * NP1 + sel] = 1;
      if (sel != N_) mask[sel] = 0.f;
    }
    __syncthreads();
  }
}

// ---------------------------------------------------------------------------
extern "C" void kernel_launch(void* const* d_in, const int* in_sizes, int n_in,
                              void* d_out, int out_size, void* d_ws, size_t ws_size,
                              hipStream_t stream) {
  const float* emb  = (const float*)d_in[0];
  const float* autm = (const float*)d_in[1];
  const float* aum  = (const float*)d_in[2];
  const float* ee   = (const float*)d_in[3];
  /* d_in[4] = temperature: unused (softmax is monotone; logits are pre-temp) */
  const int*   oen  = (const int*)d_in[5];
  const float* Wk   = (const float*)d_in[6];
  const float* bk   = (const float*)d_in[7];
  const float* Wf   = (const float*)d_in[8];
  const float* bfv  = (const float*)d_in[9];
  const float* W1   = (const float*)d_in[10];
  const float* b1   = (const float*)d_in[11];
  const float* W2   = (const float*)d_in[12];
  const float* b2   = (const float*)d_in[13];
  const float* wx   = (const float*)d_in[14];
  const float* wh   = (const float*)d_in[15];
  const float* lb   = (const float*)d_in[16];
  const float* gxg  = (const float*)d_in[17];
  const float* gxb  = (const float*)d_in[18];
  const float* ghg  = (const float*)d_in[19];
  const float* ghb  = (const float*)d_in[20];

  // Workspace layout (needs ~33.8 MB):
  //   [0, 32KB)        packed bf16 B operands for WMMA
  //   [32KB, +33.65MB) key buffer f32 [B,513,64]
  //   then             x0 f32 [B,64]
  char* ws = (char*)d_ws;
  __bf16* packedB = (__bf16*)ws;
  float*  keybuf  = (float*)(ws + 32768);
  float*  x0buf   = (float*)(ws + 32768 + (size_t)B_ * NP1 * K_ * sizeof(float));

  float* logits = (float*)d_out;                          // [T,B,513]
  int*   units  = (int*)((float*)d_out + (size_t)T_ * B_ * NP1);  // [B,513] int32

  pack_b_kernel<<<64, 256, 0, stream>>>(Wk, packedB, keybuf);
  key_gemm_kernel<<<(B_ * N_) / 128, 256, 0, stream>>>(ee, bk, packedB, keybuf);
  init_query_kernel<<<B_, 256, 0, stream>>>(emb, autm, Wf, bfv, W1, b1, W2, b2, x0buf);

  size_t smem3 = (size_t)(NP1 * K_ + 3 * H_ + 8 * H_ + 256 + NP1) * sizeof(float)
               + 256 * sizeof(int);
  (void)hipFuncSetAttribute((const void*)lstm_select_kernel,
                            hipFuncAttributeMaxDynamicSharedMemorySize, (int)smem3);
  lstm_select_kernel<<<B_, 256, smem3, stream>>>(keybuf, x0buf, aum, oen, wx, wh, lb,
                                                 gxg, gxb, ghg, ghb, logits, units);
}
